// ContrastiveWeight_53876069761119
// MI455X (gfx1250) — matrix operand order repaired and checked by reading, not compiled, not run on hardware
//
#include <hip/hip_runtime.h>
#include <hip/hip_fp16.h>

typedef __attribute__((ext_vector_type(16))) _Float16 v16h;
typedef __attribute__((ext_vector_type(8)))  _Float16 v8h;
typedef __attribute__((ext_vector_type(4)))  _Float16 v4h;
typedef __attribute__((ext_vector_type(8)))  float    v8f;
typedef __attribute__((ext_vector_type(4)))  float    v4f;

#define T_DIM   128
#define N_DIM   2048
#define D_DIM   128
#define XPITCH  136   // halves per row: 272B, 16B-aligned, conflict-free (68 dwords)
#define SPITCH  129   // floats per sim-staging row
#define TEMP_INV 5.0f // 1 / 0.2

__global__ void ContrastiveZero(float* __restrict__ out) {
    if (threadIdx.x == 0 && blockIdx.x == 0) out[0] = 0.0f;
}

__global__ __launch_bounds__(256)
void ContrastiveWeight_kernel(const float* __restrict__ in, float* __restrict__ out)
{
    __shared__ _Float16 Xh[T_DIM * XPITCH];            // ~34.8 KB, normalized f16 X_n
    __shared__ float    simbuf[8 * 16 * SPITCH];       // ~64.5 KB, per-wave row bands
    __shared__ float    blockLoss;

    const int n    = blockIdx.x;
    const int tid  = threadIdx.x;
    const int lane = tid & 31;
    const int wave = tid >> 5;

    if (tid == 0) blockLoss = 0.0f;

    // ---------- Phase 0: load + L2-normalize rows -> f16 in LDS ----------
    {
        const int row  = tid >> 1;       // 0..127
        const int hsel = tid & 1;        // which 64-element half of D
        const float* src = in + (size_t)row * (N_DIM * D_DIM) + (size_t)n * D_DIM + hsel * 64;
        float v[64];
        float ss = 0.0f;
        #pragma unroll
        for (int i = 0; i < 16; ++i) {
            v4f x = *(const v4f*)(src + i * 4);
            v[4*i+0] = x.x; v[4*i+1] = x.y; v[4*i+2] = x.z; v[4*i+3] = x.w;
            ss += x.x*x.x + x.y*x.y + x.z*x.z + x.w*x.w;
        }
        ss += __shfl_xor(ss, 1);                       // pair (2 threads per row)
        const float scale = 1.0f / fmaxf(sqrtf(ss), 1e-12f);
        _Float16* dst = &Xh[row * XPITCH + hsel * 64];
        #pragma unroll
        for (int i = 0; i < 16; ++i) {                 // 16x ds_store_b64
            v4h p = { (_Float16)(v[4*i+0]*scale), (_Float16)(v[4*i+1]*scale),
                      (_Float16)(v[4*i+2]*scale), (_Float16)(v[4*i+3]*scale) };
            *(v4h*)(dst + 4*i) = p;
        }
    }
    __syncthreads();

    // ---------- Phase 1: sim = X · X^T with v_wmma_f32_16x16x32_f16 ----------
    float* simOut = out + 1;                           // sim at out[1..]
    const int half16 = lane >> 4;                      // lane group 0/1
    const int l      = lane & 15;
    const int t0     = wave * 16;                      // this wave's row band
    const _Float16* arow = &Xh[(t0 + l) * XPITCH];

    for (int s0 = 0; s0 < 128; s0 += 16) {
        const _Float16* brow = &Xh[(s0 + l) * XPITCH]; // B column = X row (X^T)
        v8f acc = {0.f,0.f,0.f,0.f,0.f,0.f,0.f,0.f};
        #pragma unroll
        for (int k0 = 0; k0 < 128; k0 += 32) {
            // A 16x32 f16 layout: lanes<16 hold K {0..7,16..23}, lanes>=16 {8..15,24..31}
            v8h alo = *(const v8h*)(arow + k0 + 8*half16);
            v8h ahi = *(const v8h*)(arow + k0 + 16 + 8*half16);
            v16h a  = __builtin_shufflevector(alo, ahi, 0,1,2,3,4,5,6,7,8,9,10,11,12,13,14,15);
            // B 32x16 f16 layout: lanes<16 hold K {0..7,16..23}, lanes>=16 {8..15,24..31}
            v8h blo = *(const v8h*)(brow + k0 + 8*half16);
            v8h bhi = *(const v8h*)(brow + k0 + 16 + 8*half16);
            v16h b  = __builtin_shufflevector(blo, bhi, 0,1,2,3,4,5,6,7,8,9,10,11,12,13,14,15);
            acc = __builtin_amdgcn_wmma_f32_16x16x32_f16(
                      false, a, false, b, (short)0, acc, false, false);
        }
        // C/D layout: VGPR r, lanes<16 -> M=r, lanes>=16 -> M=r+8; N = lane%16
        #pragma unroll
        for (int r = 0; r < 8; ++r) {
            const int trow = t0 + r + 8*half16;
            const int s    = s0 + l;
            const float val = acc[r];
            simOut[(size_t)n * (128*128) + trow * 128 + s] = val;
            simbuf[wave * (16*SPITCH) + (r + 8*half16) * SPITCH + s] = val;
        }
    }
    // simbuf is written and read by the same wave; no block barrier needed here.

    // ---------- Phase 2: gather logits, log-softmax, loss ----------
    // Positives(t) = {c != t : c == t (mod 32)} sorted asc (always 3);
    // negatives = all c with c % 32 != t % 32, sorted asc (124).
    float* logitsOut = out + 1 + (size_t)N_DIM * 128 * 128;
    float lossLocal = 0.0f;

    for (int rr = 0; rr < 16; ++rr) {
        const int t  = t0 + rr;
        const int tq = t >> 5;   // t / 32
        const int tm = t & 31;   // t % 32
        const float* srow = &simbuf[wave * (16*SPITCH) + rr * SPITCH];

        float maxv = -1e30f;
        float sc[4];
        int   cnt = 0;
        for (int j = lane; j < 127; j += 32) {
            int col;
            if (j < 3) {
                const int i = j + (j >= tq ? 1 : 0);
                col = t + 32 * (i - tq);
            } else {
                const int m  = j - 3;
                const int b  = m / 31;
                const int r2 = m - 31 * b;
                col = 32 * b + r2 + (r2 >= tm ? 1 : 0);
            }
            const float v = srow[col];
            logitsOut[((size_t)n * 128 + t) * 127 + j] = v;   // raw gathered sim
            const float s = v * TEMP_INV;
            sc[cnt++] = s;
            maxv = fmaxf(maxv, s);
        }
        #pragma unroll
        for (int off = 16; off > 0; off >>= 1) maxv = fmaxf(maxv, __shfl_xor(maxv, off));
        float sum = 0.0f;
        for (int i2 = 0; i2 < cnt; ++i2) sum += __expf(sc[i2] - maxv);
        #pragma unroll
        for (int off = 16; off > 0; off >>= 1) sum += __shfl_xor(sum, off);
        const float logZ = maxv + __logf(sum);
        if (lane < 3) lossLocal += logZ - sc[0];   // lanes 0..2 hold positives (j=0,1,2)
    }
    #pragma unroll
    for (int off = 16; off > 0; off >>= 1) lossLocal += __shfl_xor(lossLocal, off);
    if (lane == 0) atomicAdd(&blockLoss, lossLocal);
    __syncthreads();
    if (tid == 0)
        atomicAdd(out, blockLoss * (1.0f / (128.0f * 2048.0f)));
}

extern "C" void kernel_launch(void* const* d_in, const int* in_sizes, int n_in,
                              void* d_out, int out_size, void* d_ws, size_t ws_size,
                              hipStream_t stream) {
    const float* in  = (const float*)d_in[0];
    float*       out = (float*)d_out;
    ContrastiveZero<<<1, 1, 0, stream>>>(out);                       // out[0] = loss accumulator
    ContrastiveWeight_kernel<<<N_DIM, 256, 0, stream>>>(in, out);    // one block per n
}
